// MoGHead_62569083568273
// MI455X (gfx1250) — matrix operand order repaired and checked by reading, not compiled, not run on hardware
//
#include <hip/hip_runtime.h>

typedef __bf16 bf16_t;
typedef bf16_t v16bf __attribute__((ext_vector_type(16)));
typedef bf16_t v8bf  __attribute__((ext_vector_type(8)));
typedef float  v8f   __attribute__((ext_vector_type(8)));

#define DEV __device__ __forceinline__

constexpr int LDS_STRIDE = 40;           // 64x32 tile rows padded: 80B = 20 banks
constexpr int BUFB       = 64 * LDS_STRIDE;  // elements per staging buffer (2560 bf16 = 5120B)

// ---------------------------------------------------------------------------
// Async global->LDS 16B copy (CDNA5 GLOBAL_LOAD_ASYNC_TO_LDS_B128, ASYNCcnt)
// lds_off is a byte offset from the workgroup dynamic-LDS base (== 0 here,
// kernels using this have no static LDS).
// ---------------------------------------------------------------------------
DEV void async_copy_16B(unsigned lds_off, const void* gptr) {
  asm volatile("global_load_async_to_lds_b128 %0, %1, off"
               :: "v"(lds_off), "v"(gptr) : "memory");
}
DEV void wait_async0() { asm volatile("s_wait_asynccnt 0x0" ::: "memory"); }

// ---------------------------------------------------------------------------
// WMMA fragment helpers (v_wmma_f32_16x16x32_bf16, wave32)
// A (16x32, MxK): lane L<16 -> row M=L, K = 0..15;
//                lane L>=16 -> row M=L-16, K = 16..31
// ---------------------------------------------------------------------------
DEV v16bf load_a_frag(const bf16_t* __restrict__ A, int lda, int m0, int k0, int lane) {
  const int row  = m0 + (lane & 15);
  const int koff = (lane >> 4) << 4;                 // 0 or 16
  const bf16_t* p = A + (size_t)row * lda + k0 + koff;
  v8bf lo = *(const v8bf*)(p);                       // K = k0+koff   .. +7
  v8bf hi = *(const v8bf*)(p + 8);                   // K = k0+8+koff .. +7
  v16bf a;
#pragma unroll
  for (int i = 0; i < 8; ++i) { a[i] = lo[i]; a[i + 8] = hi[i]; }
  return a;
}

// B^T (32x16) fragment from an LDS-staged 64x32 tile (row = weight row, padded stride)
// lane L<16 -> col n=L, K=0..15 ; lane L>=16 -> col n=L-16, K=16..31
DEV v16bf load_b_frag_lds(const bf16_t* sW, int t, int lane) {
  const int row = 16 * t + (lane & 15);
  const int kh  = (lane >> 4) << 4;                  // 0 or 16
  const bf16_t* p = sW + row * LDS_STRIDE + kh;
  v8bf lo = *(const v8bf*)(p);
  v8bf hi = *(const v8bf*)(p + 8);
  v16bf b;
#pragma unroll
  for (int i = 0; i < 8; ++i) { b[i] = lo[i]; b[i + 8] = hi[i]; }
  return b;
}

// fp32-weight fallback fragment (direct global, converts in-register)
DEV v16bf load_b_frag(const float* __restrict__ W, int ldb, int n0, int k0, int lane) {
  const int col = n0 + (lane & 15);
  const int kk  = k0 + ((lane >> 4) << 4);
  const float4* p = (const float4*)(W + (size_t)col * ldb + kk);
  float4 f[4];
#pragma unroll
  for (int i = 0; i < 4; ++i) f[i] = p[i];
  v16bf b;
#pragma unroll
  for (int i = 0; i < 4; ++i) {
    b[4 * i + 0] = (bf16_t)f[i].x;
    b[4 * i + 1] = (bf16_t)f[i].y;
    b[4 * i + 2] = (bf16_t)f[i].z;
    b[4 * i + 3] = (bf16_t)f[i].w;
  }
  return b;
}

DEV v8f wmma_bf16(v16bf a, v16bf b, v8f c) {
  // (neg_a, A, neg_b, B, c_mod, C, reuse_a, reuse_b)
  return __builtin_amdgcn_wmma_f32_16x16x32_bf16(false, a, false, b, (short)0, c,
                                                 false, false);
}

DEV float gelu_tanh(float x) {
  const float x3 = x * x * x;
  return 0.5f * x * (1.0f + tanhf(0.7978845608028654f * (x + 0.044715f * x3)));
}

// ---------------------------------------------------------------------------
// fp32 -> bf16 weight pre-conversion (streaming, 4 elems/thread)
// ---------------------------------------------------------------------------
__global__ __launch_bounds__(256) void cvt_f32_bf16_kernel(
    const float* __restrict__ src, bf16_t* __restrict__ dst, size_t n4) {
  const size_t i = (size_t)blockIdx.x * blockDim.x + threadIdx.x;
  if (i >= n4) return;
  const float4 f = ((const float4*)src)[i];
  bf16_t* d = dst + i * 4;
  d[0] = (bf16_t)f.x; d[1] = (bf16_t)f.y; d[2] = (bf16_t)f.z; d[3] = (bf16_t)f.w;
}

// ---------------------------------------------------------------------------
// bf16-weight GEMM, async-staged double-buffered LDS B tiles.
// C[M,N] f32 = A_bf16[M,K] @ W_bf16[N,K]^T ; K must be a power of two.
// block = 8 waves -> 128(M) x 64(N); grid = (N/64, M/128).
// dynamic LDS: 2 * 5120 B
// ---------------------------------------------------------------------------
__global__ __launch_bounds__(256) void wmma_gemm_lds_kernel(
    const bf16_t* __restrict__ A, const bf16_t* __restrict__ W,
    float* __restrict__ C, int M, int N, int K) {
  extern __shared__ bf16_t lds[];
  const int wave = threadIdx.x >> 5;
  const int lane = threadIdx.x & 31;
  const int m0 = blockIdx.y * 128 + wave * 16;
  const int n0 = blockIdx.x * 64;
  const int srow = threadIdx.x >> 2;            // 0..63  (weight row within tile)
  const int scol = (threadIdx.x & 3) * 8;       // 0,8,16,24 (K chunk)
  const unsigned sdst = (unsigned)((srow * LDS_STRIDE + scol) * 2);   // LDS byte offset
  const bf16_t* wsrc = W + (size_t)(n0 + srow) * K + scol;

  async_copy_16B(sdst, wsrc);                   // k0 = 0 -> buf 0
  wait_async0();
  __syncthreads();

  v8f acc[4] = {};
  int buf = 0;
  for (int k0 = 0; k0 < K; k0 += 32) {
    const int knext = (k0 + 32) & (K - 1);      // wrap on last iter (harmless dummy)
    async_copy_16B(sdst + (unsigned)((buf ^ 1) * BUFB * 2), wsrc + knext);
    v16bf a = load_a_frag(A, K, m0, k0, lane);
    v16bf b[4];
#pragma unroll
    for (int t = 0; t < 4; ++t) b[t] = load_b_frag_lds(lds + buf * BUFB, t, lane);
#pragma unroll
    for (int t = 0; t < 4; ++t) acc[t] = wmma_bf16(a, b[t], acc[t]);
    wait_async0();                              // this wave's stage of buf^1 done
    __syncthreads();                            // all waves done reading buf
    buf ^= 1;
  }
  // C/D layout: VGPR j, lane l -> row m0 + j + 8*(l>>4), col n0+16t+(l&15)
  const int colb  = n0 + (lane & 15);
  const int rbase = m0 + ((lane >> 4) << 3);
#pragma unroll
  for (int t = 0; t < 4; ++t)
#pragma unroll
    for (int j = 0; j < 8; ++j)
      C[(size_t)(rbase + j) * N + colb + 16 * t] = acc[t][j];
}

// ---------------------------------------------------------------------------
// bf16-weight fused gate/up GEMM + gelu-tanh epilogue, async LDS staging.
// ACT_bf16[M,N] = gelu(A@G^T) * (A@U^T) ; K power of two.
// dynamic LDS: 4 * 5120 B  (G buf0, G buf1, U buf0, U buf1)
// ---------------------------------------------------------------------------
__global__ __launch_bounds__(256) void wmma_gateup_lds_kernel(
    const bf16_t* __restrict__ A, const bf16_t* __restrict__ G,
    const bf16_t* __restrict__ U, bf16_t* __restrict__ ACT,
    int M, int N, int K) {
  extern __shared__ bf16_t lds[];
  const int wave = threadIdx.x >> 5;
  const int lane = threadIdx.x & 31;
  const int m0 = blockIdx.y * 128 + wave * 16;
  const int n0 = blockIdx.x * 64;
  const int srow = threadIdx.x >> 2;
  const int scol = (threadIdx.x & 3) * 8;
  const unsigned sdst = (unsigned)((srow * LDS_STRIDE + scol) * 2);
  const bf16_t* gsrc = G + (size_t)(n0 + srow) * K + scol;
  const bf16_t* usrc = U + (size_t)(n0 + srow) * K + scol;

  async_copy_16B(sdst, gsrc);                             // G -> buf 0 (region 0)
  async_copy_16B(sdst + (unsigned)(2 * BUFB * 2), usrc);  // U -> buf 0 (region 2)
  wait_async0();
  __syncthreads();

  v8f accg[4] = {};
  v8f accu[4] = {};
  int buf = 0;
  for (int k0 = 0; k0 < K; k0 += 32) {
    const int knext = (k0 + 32) & (K - 1);
    async_copy_16B(sdst + (unsigned)((buf ^ 1) * BUFB * 2), gsrc + knext);
    async_copy_16B(sdst + (unsigned)((2 + (buf ^ 1)) * BUFB * 2), usrc + knext);
    v16bf a = load_a_frag(A, K, m0, k0, lane);
    v16bf bg[4], bu[4];
#pragma unroll
    for (int t = 0; t < 4; ++t) {
      bg[t] = load_b_frag_lds(lds + buf * BUFB, t, lane);
      bu[t] = load_b_frag_lds(lds + (2 + buf) * BUFB, t, lane);
    }
#pragma unroll
    for (int t = 0; t < 4; ++t) {
      accg[t] = wmma_bf16(a, bg[t], accg[t]);
      accu[t] = wmma_bf16(a, bu[t], accu[t]);
    }
    wait_async0();
    __syncthreads();
    buf ^= 1;
  }
  const int rbase = m0 + ((lane >> 4) << 3);
#pragma unroll
  for (int t = 0; t < 4; ++t) {
    const int col = n0 + 16 * t + (lane & 15);
#pragma unroll
    for (int j = 0; j < 8; ++j) {
      float v = gelu_tanh(accg[t][j]) * accu[t][j];
      ACT[(size_t)(rbase + j) * N + col] = (bf16_t)v;
    }
  }
}

// ---------------------------------------------------------------------------
// fp32-weight fallback GEMMs (direct global loads + in-register cvt)
// ---------------------------------------------------------------------------
__global__ __launch_bounds__(256) void wmma_gemm_f32w_kernel(
    const bf16_t* __restrict__ A, const float* __restrict__ W,
    float* __restrict__ C, int M, int N, int K) {
  const int wave = threadIdx.x >> 5;
  const int lane = threadIdx.x & 31;
  const int m0 = blockIdx.y * 128 + wave * 16;
  const int n0 = blockIdx.x * 64;
  v8f acc[4] = {};
  for (int k0 = 0; k0 < K; k0 += 32) {
    v16bf a = load_a_frag(A, K, m0, k0, lane);
#pragma unroll
    for (int t = 0; t < 4; ++t) {
      v16bf b = load_b_frag(W, K, n0 + 16 * t, k0, lane);
      acc[t] = wmma_bf16(a, b, acc[t]);
    }
  }
  const int colb  = n0 + (lane & 15);
  const int rbase = m0 + ((lane >> 4) << 3);
#pragma unroll
  for (int t = 0; t < 4; ++t)
#pragma unroll
    for (int j = 0; j < 8; ++j)
      C[(size_t)(rbase + j) * N + colb + 16 * t] = acc[t][j];
}

__global__ __launch_bounds__(256) void wmma_gateup_f32w_kernel(
    const bf16_t* __restrict__ A, const float* __restrict__ G,
    const float* __restrict__ U, bf16_t* __restrict__ ACT,
    int M, int N, int K) {
  const int wave = threadIdx.x >> 5;
  const int lane = threadIdx.x & 31;
  const int m0 = blockIdx.y * 128 + wave * 16;
  const int n0 = blockIdx.x * 64;
  v8f accg[4] = {};
  v8f accu[4] = {};
  for (int k0 = 0; k0 < K; k0 += 32) {
    v16bf a = load_a_frag(A, K, m0, k0, lane);
#pragma unroll
    for (int t = 0; t < 4; ++t) {
      v16bf bg = load_b_frag(G, K, n0 + 16 * t, k0, lane);
      accg[t] = wmma_bf16(a, bg, accg[t]);
      v16bf bu = load_b_frag(U, K, n0 + 16 * t, k0, lane);
      accu[t] = wmma_bf16(a, bu, accu[t]);
    }
  }
  const int rbase = m0 + ((lane >> 4) << 3);
#pragma unroll
  for (int t = 0; t < 4; ++t) {
    const int col = n0 + 16 * t + (lane & 15);
#pragma unroll
    for (int j = 0; j < 8; ++j) {
      float v = gelu_tanh(accg[t][j]) * accu[t][j];
      ACT[(size_t)(rbase + j) * N + col] = (bf16_t)v;
    }
  }
}

// ---------------------------------------------------------------------------
// Elementwise / reduction kernels
// ---------------------------------------------------------------------------
DEV float block_reduce_sum(float v, float* sh) {
  const int lane = threadIdx.x & 31;
  const int wave = threadIdx.x >> 5;
#pragma unroll
  for (int d = 16; d > 0; d >>= 1) v += __shfl_down(v, d, 32);
  if (lane == 0) sh[wave] = v;
  __syncthreads();
  float r = (threadIdx.x < (blockDim.x >> 5)) ? sh[threadIdx.x] : 0.0f;
  if (wave == 0) {
#pragma unroll
    for (int d = 16; d > 0; d >>= 1) r += __shfl_down(r, d, 32);
    if (lane == 0) sh[0] = r;
  }
  __syncthreads();
  return sh[0];
}

// y = rmsnorm(x, w) -> bf16 (and optional f32 copy)
__global__ __launch_bounds__(256) void rmsnorm_kernel(
    const float* __restrict__ X, const float* __restrict__ w,
    bf16_t* __restrict__ Ybf, float* __restrict__ Yf, int H) {
  __shared__ float sh[8];
  const size_t base = (size_t)blockIdx.x * H;
  float ssq = 0.f;
  for (int i = threadIdx.x; i < H; i += blockDim.x) {
    float v = X[base + i];
    ssq += v * v;
  }
  const float tot = block_reduce_sum(ssq, sh);
  const float scale = rsqrtf(tot / (float)H + 1e-6f);
  for (int i = threadIdx.x; i < H; i += blockDim.x) {
    float v = X[base + i] * scale * (1.0f + w[i]);
    Ybf[base + i] = (bf16_t)v;
    if (Yf) Yf[base + i] = v;
  }
}

// h += rmsnorm(tmp, post_w)
__global__ __launch_bounds__(256) void residual_post_kernel(
    float* __restrict__ h, const float* __restrict__ T,
    const float* __restrict__ w, int H) {
  __shared__ float sh[8];
  const size_t base = (size_t)blockIdx.x * H;
  float ssq = 0.f;
  for (int i = threadIdx.x; i < H; i += blockDim.x) {
    float v = T[base + i];
    ssq += v * v;
  }
  const float tot = block_reduce_sum(ssq, sh);
  const float scale = rsqrtf(tot / (float)H + 1e-6f);
  for (int i = threadIdx.x; i < H; i += blockDim.x)
    h[base + i] += T[base + i] * scale * (1.0f + w[i]);
}

// idx[t] = argmax_n(logits + gumbel)  (LSE of log_softmax is per-token const)
__global__ __launch_bounds__(256) void routing_kernel(
    const float* __restrict__ logits, const float* __restrict__ gu,
    int* __restrict__ idx, int BT) {
  const int t = blockIdx.x * blockDim.x + threadIdx.x;
  if (t >= BT) return;
  const float* l = logits + (size_t)t * 64;
  const float* u = gu + (size_t)t * 64;
  float best = -3.4e38f;
  int bi = 0;
  for (int n = 0; n < 64; ++n) {
    float g = -logf(-logf(u[n] + 1e-8f) + 1e-8f);
    float s = l[n] + g;
    if (s > best) { best = s; bi = n; }   // strict > keeps first max (argmax)
  }
  idx[t] = bi;
}

// mu[t,r] = sum_h hF[t,h] * w_mus[(idx*R+r)*H + h]; one block per token
__global__ __launch_bounds__(256) void mu_kernel(
    const float* __restrict__ hF, const float* __restrict__ w_mus,
    const int* __restrict__ idx, float* __restrict__ mu) {
  const int t = blockIdx.x;
  const int lane = threadIdx.x & 31;
  const int wave = threadIdx.x >> 5;
  const int comp = idx[t];
  const float* h = hF + (size_t)t * 2048;
  for (int r = wave; r < 64; r += 8) {
    const float* wr = w_mus + ((size_t)comp * 64 + r) * 2048;
    float s = 0.f;
    for (int i = lane; i < 2048; i += 32) s += h[i] * wr[i];
#pragma unroll
    for (int d = 16; d > 0; d >>= 1) s += __shfl_down(s, d, 32);
    if (lane == 0) mu[(size_t)t * 64 + r] = s;
  }
}

// out1 = (low_mat[idx] @ mu) * exp(logs) + mu_res ; out2 = logs
__global__ __launch_bounds__(256) void final_kernel(
    const float* __restrict__ hF, const float* __restrict__ w_logs,
    const float* __restrict__ low_mat, const float* __restrict__ mu,
    const float* __restrict__ mu_res, const int* __restrict__ idx,
    float* __restrict__ out1, float* __restrict__ out2) {
  __shared__ float sh[8];
  __shared__ float smu[64];
  const int t = blockIdx.x;
  if (threadIdx.x < 64) smu[threadIdx.x] = mu[(size_t)t * 64 + threadIdx.x];
  float s = 0.f;
  for (int i = threadIdx.x; i < 2048; i += 256)
    s += hF[(size_t)t * 2048 + i] * w_logs[i];
  const float tot = block_reduce_sum(s, sh);     // barriers also cover smu
  const float lg = fmaxf(tot, -4.0f);            // MIN_LOG_STD
  if (threadIdx.x == 0) out2[t] = lg;
  const float els = expf(lg);
  const int comp = idx[t];
  for (int o = threadIdx.x; o < 512; o += 256) {
    const float* lm = low_mat + ((size_t)comp * 512 + o) * 64;
    float acc = 0.f;
#pragma unroll
    for (int r = 0; r < 64; ++r) acc += lm[r] * smu[r];
    out1[(size_t)t * 512 + o] = acc * els + mu_res[(size_t)t * 512 + o];
  }
}

// ---------------------------------------------------------------------------
// Host orchestration
// ---------------------------------------------------------------------------
static inline void cvt_weights(const float* src, bf16_t* dst, size_t n,
                               hipStream_t stream) {
  const size_t n4 = n / 4;
  const int blocks = (int)((n4 + 255) / 256);
  cvt_f32_bf16_kernel<<<blocks, 256, 0, stream>>>(src, dst, n4);
}

extern "C" void kernel_launch(void* const* d_in, const int* in_sizes, int n_in,
                              void* d_out, int out_size, void* d_ws, size_t ws_size,
                              hipStream_t stream) {
  (void)in_sizes; (void)n_in; (void)out_size;
  constexpr int BT = 4096, H = 2048, I = 8192, L = 2, N = 64, OUT = 512;
  constexpr size_t LDS1 = 2 * BUFB * sizeof(bf16_t);   // gemm dynamic LDS
  constexpr size_t LDS2 = 4 * BUFB * sizeof(bf16_t);   // gateup dynamic LDS

  const float* x        = (const float*)d_in[0];
  const float* gumbel_u = (const float*)d_in[1];
  const float* pre_w    = (const float*)d_in[2];
  const float* gate_w   = (const float*)d_in[3];
  const float* up_w     = (const float*)d_in[4];
  const float* down_w   = (const float*)d_in[5];
  const float* post_w   = (const float*)d_in[6];
  const float* final_w  = (const float*)d_in[7];
  const float* w_logits = (const float*)d_in[8];
  const float* w_mus    = (const float*)d_in[9];
  const float* w_logs   = (const float*)d_in[10];
  const float* w_else   = (const float*)d_in[11];
  const float* low_mat  = (const float*)d_in[12];

  char* ws = (char*)d_ws;
  float*  h      = (float*)ws;   ws += (size_t)BT * H * 4;
  bf16_t* y_bf   = (bf16_t*)ws;  ws += (size_t)BT * H * 2;
  bf16_t* act_bf = (bf16_t*)ws;  ws += (size_t)BT * I * 2;
  float*  tmp    = (float*)ws;   ws += (size_t)BT * H * 4;   // also hF after final norm
  float*  logits = (float*)ws;   ws += (size_t)BT * N * 4;
  float*  mu     = (float*)ws;   ws += (size_t)BT * 64 * 4;
  float*  mu_res = (float*)ws;   ws += (size_t)BT * OUT * 4;
  int*    idx    = (int*)ws;     ws += (size_t)BT * 4;

  // Optional bf16 weight staging (gated on available workspace)
  const size_t base_bytes = (size_t)(ws - (char*)d_ws);
  const size_t wbf_bytes  = ((size_t)L * I * H * 3 + (size_t)N * H + (size_t)OUT * H) * 2;
  const bool use_bf16w = (ws_size >= base_bytes + wbf_bytes);
  bf16_t* gate_bf = nullptr; bf16_t* up_bf = nullptr; bf16_t* down_bf = nullptr;
  bf16_t* wlog_bf = nullptr; bf16_t* wels_bf = nullptr;
  if (use_bf16w) {
    gate_bf = (bf16_t*)ws;  ws += (size_t)L * I * H * 2;
    up_bf   = (bf16_t*)ws;  ws += (size_t)L * I * H * 2;
    down_bf = (bf16_t*)ws;  ws += (size_t)L * H * I * 2;
    wlog_bf = (bf16_t*)ws;  ws += (size_t)N * H * 2;
    wels_bf = (bf16_t*)ws;  ws += (size_t)OUT * H * 2;
    cvt_weights(gate_w,   gate_bf, (size_t)L * I * H, stream);
    cvt_weights(up_w,     up_bf,   (size_t)L * I * H, stream);
    cvt_weights(down_w,   down_bf, (size_t)L * H * I, stream);
    cvt_weights(w_logits, wlog_bf, (size_t)N * H,     stream);
    cvt_weights(w_else,   wels_bf, (size_t)OUT * H,   stream);
  }

  hipMemcpyAsync(h, x, (size_t)BT * H * 4, hipMemcpyDeviceToDevice, stream);

  for (int l = 0; l < L; ++l) {
    rmsnorm_kernel<<<BT, 256, 0, stream>>>(h, pre_w + (size_t)l * H, y_bf, nullptr, H);
    if (use_bf16w) {
      wmma_gateup_lds_kernel<<<dim3(I / 64, BT / 128), 256, LDS2, stream>>>(
          y_bf, gate_bf + (size_t)l * I * H, up_bf + (size_t)l * I * H, act_bf, BT, I, H);
      wmma_gemm_lds_kernel<<<dim3(H / 64, BT / 128), 256, LDS1, stream>>>(
          act_bf, down_bf + (size_t)l * H * I, tmp, BT, H, I);
    } else {
      wmma_gateup_f32w_kernel<<<dim3(I / 64, BT / 128), 256, 0, stream>>>(
          y_bf, gate_w + (size_t)l * I * H, up_w + (size_t)l * I * H, act_bf, BT, I, H);
      wmma_gemm_f32w_kernel<<<dim3(H / 64, BT / 128), 256, 0, stream>>>(
          act_bf, down_w + (size_t)l * H * I, tmp, BT, H, I);
    }
    residual_post_kernel<<<BT, 256, 0, stream>>>(h, tmp, post_w + (size_t)l * H, H);
  }

  rmsnorm_kernel<<<BT, 256, 0, stream>>>(h, final_w, y_bf, tmp, H);  // tmp = hF (f32)

  if (use_bf16w) {
    wmma_gemm_lds_kernel<<<dim3(N / 64, BT / 128), 256, LDS1, stream>>>(
        y_bf, wlog_bf, logits, BT, N, H);
    wmma_gemm_lds_kernel<<<dim3(OUT / 64, BT / 128), 256, LDS1, stream>>>(
        y_bf, wels_bf, mu_res, BT, OUT, H);
  } else {
    wmma_gemm_f32w_kernel<<<dim3(N / 64, BT / 128), 256, 0, stream>>>(
        y_bf, w_logits, logits, BT, N, H);
    wmma_gemm_f32w_kernel<<<dim3(OUT / 64, BT / 128), 256, 0, stream>>>(
        y_bf, w_else, mu_res, BT, OUT, H);
  }
  routing_kernel<<<BT / 256, 256, 0, stream>>>(logits, gumbel_u, idx, BT);
  mu_kernel<<<BT, 256, 0, stream>>>(tmp, w_mus, idx, mu);

  float* out1 = (float*)d_out;
  float* out2 = out1 + (size_t)BT * OUT;
  final_kernel<<<BT, 256, 0, stream>>>(tmp, w_logs, low_mat, mu, mu_res, idx, out1, out2);
}
